// OptimizedLSTM_52286931861807
// MI455X (gfx1250) — matrix operand order repaired and checked by reading, not compiled, not run on hardware
//
#include <hip/hip_runtime.h>
#include <hip/hip_bf16.h>
#include <stdint.h>

// ---------------------------------------------------------------------------
// LSTM for MI455X / gfx1250.  B=64, S=512, D=512, H=1024, 4H=4096, OUT=5.
// Pipeline:
//   1) prep:        W_ih,W_hh -> bf16, bias=b_ih+b_hh, zero h/c/ctr
//   2) gates_x:     big parallel bf16 WMMA GEMM (gather-A from embedding)
//   3) lstm_scan:   16 persistent WGs; double-buffered TDM gate-tile
//                   prefetch (next step streams while current step's WMMA
//                   GEMM runs), W_hh bf16 resident in L2, in-register cell
//                   update, atomic device barrier per step
//   4) out:         tiny [64x1024]x[1024x5] GEMV
// ---------------------------------------------------------------------------

typedef __bf16 bf16_t;
typedef bf16_t v16bf __attribute__((ext_vector_type(16)));
typedef bf16_t v8bf  __attribute__((ext_vector_type(8)));
typedef float  v8f   __attribute__((ext_vector_type(8)));
typedef unsigned int v4u __attribute__((ext_vector_type(4)));
typedef int    v8i   __attribute__((ext_vector_type(8)));
typedef int    v4i   __attribute__((ext_vector_type(4)));

constexpr int B_ = 64, S_ = 512, D_ = 512, H_ = 1024, G_ = 4096, OUT_ = 5;

// workspace layout (bytes)
constexpr size_t WS_CTR  = 0;
constexpr size_t WS_BIAS = 256;                                   // 4096 f32
constexpr size_t WS_WIH  = WS_BIAS + (size_t)G_ * 4;              // 2M bf16
constexpr size_t WS_WHH  = WS_WIH  + (size_t)G_ * D_ * 2;         // 4M bf16
constexpr size_t WS_HB   = WS_WHH  + (size_t)G_ * H_ * 2;         // 64K bf16
constexpr size_t WS_GX   = WS_HB   + (size_t)B_ * H_ * 2;         // S*B*4H bf16

// ---------------------------------------------------------------------------
// 1) prep: weight conversion, bias fuse, state/barrier init
// ---------------------------------------------------------------------------
__global__ void prep_kernel(const float* __restrict__ Wih,
                            const float* __restrict__ Whh,
                            const float* __restrict__ bih,
                            const float* __restrict__ bhh,
                            bf16_t* wih_b, bf16_t* whh_b, float* bias,
                            float* h_out, float* c_out, bf16_t* h_b,
                            unsigned int* ctr) {
  size_t i = (size_t)blockIdx.x * blockDim.x + threadIdx.x;
  const size_t NIH = (size_t)G_ * D_;
  const size_t NHH = (size_t)G_ * H_;
  if (i < NIH) { wih_b[i] = (bf16_t)Wih[i]; return; }
  i -= NIH;
  if (i < NHH) { whh_b[i] = (bf16_t)Whh[i]; return; }
  i -= NHH;
  if (i < (size_t)G_) { bias[i] = bih[i] + bhh[i]; return; }
  i -= G_;
  if (i < (size_t)B_ * H_) {
    h_out[i] = 0.f; c_out[i] = 0.f; h_b[i] = (bf16_t)0.f; return;
  }
  i -= (size_t)B_ * H_;
  if (i == 0) *ctr = 0u;
}

// ---------------------------------------------------------------------------
// 2) gates_x[s,b,g] = emb[x[b,s]] . W_ih[g,:] + (b_ih+b_hh)[g], stored bf16
//    Block tile: 64 rows x 128 gate-cols, 8 waves, 4 C-tiles per wave.
// ---------------------------------------------------------------------------
__global__ __launch_bounds__(256)
void gates_x_kernel(const int* __restrict__ x, const float* __restrict__ emb,
                    const bf16_t* __restrict__ wih_b,
                    const float* __restrict__ bias, bf16_t* __restrict__ gx) {
  constexpr int LDA = 40;                 // padded LDS row stride (bf16 elems)
  __shared__ bf16_t At[64 * LDA];
  const int cb   = blockIdx.x;            // 0..31  gate-col block (128 wide)
  const int rb   = blockIdx.y;            // 0..511 row block (64 rows)
  const int tid  = threadIdx.x;
  const int lane = tid & 31;
  const int wv   = tid >> 5;              // 8 waves
  const int mtile = wv >> 1;              // 0..3
  const int ng    = wv & 1;               // col half
  const int ln15  = lane & 15;
  const int lhalf = lane >> 4;

  // per-thread A staging coordinates (gathered embedding rows)
  const int lr   = tid >> 2;              // LDS row 0..63
  const int kc0  = (tid & 3) * 8;         // 8 elems per thread per chunk
  const int grow = rb * 64 + lr;          // global A row = s*64 + b
  const int s    = grow >> 6;
  const int b    = grow & 63;
  const float* erow = emb + (size_t)x[b * S_ + s] * D_;

  v8f acc[4] = {};
  for (int kc = 0; kc < D_ / 32; ++kc) {
    #pragma unroll
    for (int u = 0; u < 8; ++u)
      At[lr * LDA + kc0 + u] = (bf16_t)erow[kc * 32 + kc0 + u];
    __syncthreads();

    // A fragment (contiguous K, matching B): lanes<16 -> K{0..15},
    // lanes>=16 -> K{16..31}, row M = lane&15 within tile.
    const bf16_t* ap = &At[(mtile * 16 + ln15) * LDA + (lhalf ? 16 : 0)];
    union { v16bf v; v8bf h[2]; } a;
    a.h[0] = *(const v8bf*)(ap);
    a.h[1] = *(const v8bf*)(ap + 8);

    #pragma unroll
    for (int t = 0; t < 4; ++t) {
      const int gcol = cb * 128 + (ng * 4 + t) * 16 + ln15;
      // B fragment: 16 contiguous bf16 from W_ih row gcol (no transpose)
      const bf16_t* bp = wih_b + (size_t)gcol * D_ + kc * 32 + (lhalf << 4);
      v16bf bfr = *(const v16bf*)bp;
      acc[t] = __builtin_amdgcn_wmma_f32_16x16x32_bf16(
          false, a.v, false, bfr, (short)0, acc[t], false, false);
    }
    __syncthreads();
  }

  #pragma unroll
  for (int t = 0; t < 4; ++t) {
    const int gcol = cb * 128 + (ng * 4 + t) * 16 + ln15;
    const float bc = bias[gcol];
    #pragma unroll
    for (int j = 0; j < 8; ++j) {
      const int row = rb * 64 + mtile * 16 + (lhalf << 3) + j;
      gx[(size_t)row * G_ + gcol] = (bf16_t)(acc[t][j] + bc);
    }
  }
}

// ---------------------------------------------------------------------------
// TDM: 2-D tile (bf16) global -> LDS.  D# per CDNA5 ISA.
// This toolchain exposes the 6-arg builtin form:
//   (u32x4 g0, i32x8 g1, i32x4 g2, i32x4 g3, i32x8 pad, i32 cpol)
// ---------------------------------------------------------------------------
__device__ __forceinline__ void tdm_load_tile_2d(uint32_t lds_byte, uint64_t ga,
                                                 uint32_t tdim0, uint32_t tdim1,
                                                 uint32_t tile0, uint32_t tile1,
                                                 uint32_t stride0) {
  v4u g0;
  g0.x = 1u;                                            // count=1 (valid, user)
  g0.y = lds_byte;                                      // lds_addr
  g0.z = (uint32_t)(ga & 0xffffffffull);                // global_addr[31:0]
  g0.w = (uint32_t)((ga >> 32) & 0x1ffffffull) | (2u << 30);  // [56:32] | type=2
  v8i g1;
  g1[0] = (int)(1u << 16);                              // data_size=1 (2 bytes)
  g1[1] = (int)((tdim0 & 0xffffu) << 16);               // tensor_dim0 lo
  g1[2] = (int)(((tdim0 >> 16) & 0xffffu) | ((tdim1 & 0xffffu) << 16));
  g1[3] = (int)(((tdim1 >> 16) & 0xffffu) | ((tile0 & 0xffffu) << 16));
  g1[4] = (int)(tile1 & 0xffffu);                       // tile_dim1, tile_dim2=0
  g1[5] = (int)stride0;                                 // tensor_dim0_stride lo
  g1[6] = 0;
  g1[7] = 0;
  v4i z4 = {0, 0, 0, 0};
  v8i z8 = {0, 0, 0, 0, 0, 0, 0, 0};
  __builtin_amdgcn_tensor_load_to_lds(g0, g1, z4, z4, z8, 0);
}

// issue the 4 gate-section tiles of step s into LDS buffer `buf`
__device__ __forceinline__ void issue_gx_tiles(const bf16_t* gx, int s, int j0,
                                               uint32_t lds_base, int buf) {
  const uint32_t boff = lds_base + (uint32_t)buf * (4 * 64 * 64 * 2);
  #pragma unroll
  for (int sct = 0; sct < 4; ++sct) {
    uint64_t ga = (uint64_t)(uintptr_t)(gx + (size_t)s * B_ * G_ +
                                        (size_t)sct * H_ + j0);
    tdm_load_tile_2d(boff + (uint32_t)sct * (64 * 64 * 2), ga,
                     (uint32_t)G_, (uint32_t)B_, 64u, 64u, (uint32_t)G_);
  }
}

// ---------------------------------------------------------------------------
// 3) sequential scan: 16 persistent WGs; WG n owns hidden cols [n*64, n*64+64)
//    and computes all 4 gate sections for that slice -> cell update is
//    entirely in-register per lane.  TDM gate tiles are double-buffered:
//    step s+1 streams from HBM while step s's WMMA GEMM executes.
// ---------------------------------------------------------------------------
__global__ __launch_bounds__(256)
void lstm_scan_kernel(const bf16_t* __restrict__ gx,
                      const bf16_t* __restrict__ whh_b,
                      bf16_t* __restrict__ h_b,
                      float* __restrict__ h_out, float* __restrict__ c_out,
                      unsigned int* ctr) {
  __shared__ bf16_t gxs[2 * 4 * 64 * 64];  // 64 KB: double-buffered tiles
  const int wgid  = blockIdx.x;            // 0..15
  const int nwg   = gridDim.x;
  const int tid   = threadIdx.x;
  const int lane  = tid & 31;
  const int wv    = tid >> 5;              // 8 waves
  const int mtile = wv >> 1;               // batch tile 0..3
  const int jg    = wv & 1;                // column half of slice (32 cols)
  const int j0    = wgid * 64;             // hidden slice base
  const int ln15  = lane & 15;
  const int lhalf = lane >> 4;

  const uint32_t lds_base = (uint32_t)(uintptr_t)(&gxs[0]);

  // prologue: prefetch step 0 tiles into buffer 0
  if (tid == 0) issue_gx_tiles(gx, 0, j0, lds_base, 0);

  for (int s = 0; s < S_; ++s) {
    const int cur = s & 1;
    // ---- prefetch next step into the other buffer, then retire current ----
    if (tid == 0) {
      if (s + 1 < S_) {
        issue_gx_tiles(gx, s + 1, j0, lds_base, cur ^ 1);
        __builtin_amdgcn_s_wait_tensorcnt(4);   // 4 oldest (current buf) done
      } else {
        __builtin_amdgcn_s_wait_tensorcnt(0);
      }
    }
    __syncthreads();
    const bf16_t* gxc = &gxs[cur * (4 * 64 * 64)];

    // ---- recurrent GEMM: acc[sct*2+q] covers cols jg*32+q*16+ln15 ----
    v8f acc[8] = {};
    for (int kc = 0; kc < H_ / 32; ++kc) {
      const bf16_t* ap = h_b + (size_t)(mtile * 16 + ln15) * H_ +
                         kc * 32 + (lhalf ? 16 : 0);
      union { v16bf v; v8bf h[2]; } a;
      a.h[0] = *(const v8bf*)(ap);
      a.h[1] = *(const v8bf*)(ap + 8);
      #pragma unroll
      for (int sct = 0; sct < 4; ++sct) {
        #pragma unroll
        for (int q = 0; q < 2; ++q) {
          const int gcol = sct * H_ + j0 + jg * 32 + q * 16 + ln15;
          const bf16_t* bp = whh_b + (size_t)gcol * H_ + kc * 32 + (lhalf << 4);
          v16bf bfr = *(const v16bf*)bp;
          acc[sct * 2 + q] = __builtin_amdgcn_wmma_f32_16x16x32_bf16(
              false, a.v, false, bfr, (short)0, acc[sct * 2 + q], false, false);
        }
      }
    }

    // ---- cell update, fully in-register per lane ----
    #pragma unroll
    for (int q = 0; q < 2; ++q) {
      const int jl = jg * 32 + q * 16 + ln15;          // 0..63 within slice
      #pragma unroll
      for (int j = 0; j < 8; ++j) {
        const int b  = mtile * 16 + lhalf * 8 + j;
        const float gi = acc[0 + q][j] + (float)gxc[(0 * 64 + b) * 64 + jl];
        const float gf = acc[2 + q][j] + (float)gxc[(1 * 64 + b) * 64 + jl];
        const float gg = acc[4 + q][j] + (float)gxc[(2 * 64 + b) * 64 + jl];
        const float go = acc[6 + q][j] + (float)gxc[(3 * 64 + b) * 64 + jl];
        const float iv = 1.f / (1.f + __expf(-gi));
        const float fv = 1.f / (1.f + __expf(-gf));
        const float gv = tanhf(gg);
        const float ov = 1.f / (1.f + __expf(-go));
        const size_t hidx = (size_t)b * H_ + j0 + jl;
        const float cv = fv * c_out[hidx] + iv * gv;
        const float hv = ov * tanhf(cv);
        c_out[hidx] = cv;
        h_out[hidx] = hv;
        h_b[hidx]   = (bf16_t)hv;
      }
    }

    // ---- device-wide barrier (sense = monotonically increasing count) ----
    // Also fences the back buffer: TDM writes for step s+1 were issued by
    // this WG only, and every WG finished reading its own buffers above.
    __threadfence();
    __syncthreads();
    if (tid == 0) {
      __hip_atomic_fetch_add(ctr, 1u, __ATOMIC_RELEASE, __HIP_MEMORY_SCOPE_AGENT);
      const unsigned int target = (unsigned int)nwg * (unsigned int)(s + 1);
      while (__hip_atomic_load(ctr, __ATOMIC_ACQUIRE,
                               __HIP_MEMORY_SCOPE_AGENT) < target)
        __builtin_amdgcn_s_sleep(2);
    }
    __syncthreads();
    __threadfence();
  }
}

// ---------------------------------------------------------------------------
// 4) out = h @ W_out^T + b_out   (64x5, trivial)
// ---------------------------------------------------------------------------
__global__ void out_kernel(const float* __restrict__ h_out,
                           const float* __restrict__ Wout,
                           const float* __restrict__ bout,
                           float* __restrict__ out) {
  const int t = threadIdx.x;
  if (t >= B_ * OUT_) return;
  const int b = t / OUT_, o = t % OUT_;
  const float* hr = h_out + (size_t)b * H_;
  const float* wr = Wout + (size_t)o * H_;
  float acc = 0.f;
  for (int k = 0; k < H_; ++k) acc += hr[k] * wr[k];
  out[t] = acc + bout[o];
}

// ---------------------------------------------------------------------------
extern "C" void kernel_launch(void* const* d_in, const int* in_sizes, int n_in,
                              void* d_out, int out_size, void* d_ws,
                              size_t ws_size, hipStream_t stream) {
  const int*   x    = (const int*)d_in[0];
  const float* emb  = (const float*)d_in[1];
  const float* Wih  = (const float*)d_in[2];
  const float* bih  = (const float*)d_in[3];
  const float* Whh  = (const float*)d_in[4];
  const float* bhh  = (const float*)d_in[5];
  const float* Wout = (const float*)d_in[6];
  const float* bout = (const float*)d_in[7];
  (void)in_sizes; (void)n_in; (void)out_size; (void)ws_size;

  char* ws = (char*)d_ws;
  unsigned int* ctr = (unsigned int*)(ws + WS_CTR);
  float*  bias  = (float*)(ws + WS_BIAS);
  bf16_t* wih_b = (bf16_t*)(ws + WS_WIH);
  bf16_t* whh_b = (bf16_t*)(ws + WS_WHH);
  bf16_t* h_b   = (bf16_t*)(ws + WS_HB);
  bf16_t* gx    = (bf16_t*)(ws + WS_GX);

  float* out   = (float*)d_out;            // [64,5]
  float* h_out = out + B_ * OUT_;          // [64,1024]
  float* c_out = h_out + (size_t)B_ * H_;  // [64,1024]

  {
    size_t total = (size_t)G_ * D_ + (size_t)G_ * H_ + G_ + (size_t)B_ * H_ + 1;
    int blocks = (int)((total + 255) / 256);
    prep_kernel<<<blocks, 256, 0, stream>>>(Wih, Whh, bih, bhh, wih_b, whh_b,
                                            bias, h_out, c_out, h_b, ctr);
  }
  {
    dim3 grid(G_ / 128, (S_ * B_) / 64);   // 32 x 512 blocks
    gates_x_kernel<<<grid, 256, 0, stream>>>(x, emb, wih_b, bias, gx);
  }
  lstm_scan_kernel<<<16, 256, 0, stream>>>(gx, whh_b, h_b, h_out, c_out, ctr);
  out_kernel<<<1, 512, 0, stream>>>(h_out, Wout, bout, out);
}
